// TernaryLinear_84559316124170
// MI455X (gfx1250) — 8-bit matrix operand order repaired and checked by reading, compiled, not run on hardware
//
#include <hip/hip_runtime.h>
#include <stdint.h>

#define M_DIM 8192
#define K_DIM 4096
#define N_DIM 4096

#define PART_OFF   ((size_t)0)
#define WSCALE_OFF ((size_t)4096)
#define WQ_OFF     ((size_t)8192)
#define XQ_OFF     (WQ_OFF + (size_t)N_DIM * K_DIM)
#define DQ_OFF     (XQ_OFF + (size_t)M_DIM * K_DIM)

typedef __attribute__((ext_vector_type(8))) int   v8i;
typedef __attribute__((ext_vector_type(4))) int   v4i;
typedef __attribute__((ext_vector_type(2))) int   v2i;
typedef __attribute__((ext_vector_type(4))) float v4f;

union FragA { v8i v; v2i h[4]; };
union FragB { v8i v; v4i q[2]; };

__global__ __launch_bounds__(256) void k_abs_partial(const float* __restrict__ W,
                                                     float* __restrict__ part) {
    __shared__ float red[256];
    const v4f* W4 = (const v4f*)W;
    const size_t total = (size_t)N_DIM * K_DIM / 4;
    float s = 0.f;
    for (size_t i = (size_t)blockIdx.x * 256 + threadIdx.x; i < total;
         i += (size_t)256 * 1024) {
        v4f w = W4[i];
        s += fabsf(w.x) + fabsf(w.y) + fabsf(w.z) + fabsf(w.w);
    }
    red[threadIdx.x] = s;
    __syncthreads();
    for (int o = 128; o > 0; o >>= 1) {
        if ((int)threadIdx.x < o) red[threadIdx.x] += red[threadIdx.x + o];
        __syncthreads();
    }
    if (threadIdx.x == 0) part[blockIdx.x] = red[0];
}

__global__ __launch_bounds__(256) void k_abs_final(const float* __restrict__ part,
                                                   float* __restrict__ wscale) {
    __shared__ float red[256];
    int t = threadIdx.x;
    float s = part[t] + part[t + 256] + part[t + 512] + part[t + 768];
    red[t] = s;
    __syncthreads();
    for (int o = 128; o > 0; o >>= 1) {
        if (t < o) red[t] += red[t + o];
        __syncthreads();
    }
    if (t == 0) *wscale = red[0] / ((float)N_DIM * (float)K_DIM);
}

__global__ __launch_bounds__(256) void k_quant_w(const float* __restrict__ W,
                                                 const float* __restrict__ wscale,
                                                 uint8_t* __restrict__ Wq) {
    size_t i = (size_t)blockIdx.x * 256 + threadIdx.x;
    float inv = 1.0f / (*wscale + 1e-10f);
    v4f w = ((const v4f*)W)[i];
    int q0 = (int)fminf(1.f, fmaxf(-1.f, rintf(w.x * inv)));
    int q1 = (int)fminf(1.f, fmaxf(-1.f, rintf(w.y * inv)));
    int q2 = (int)fminf(1.f, fmaxf(-1.f, rintf(w.z * inv)));
    int q3 = (int)fminf(1.f, fmaxf(-1.f, rintf(w.w * inv)));
    uint32_t p = (uint32_t)(q0 & 0xFF) | ((uint32_t)(q1 & 0xFF) << 8) |
                 ((uint32_t)(q2 & 0xFF) << 16) | ((uint32_t)(q3 & 0xFF) << 24);
    ((uint32_t*)Wq)[i] = p;
}

__global__ __launch_bounds__(256) void k_quant_x(const float* __restrict__ X,
                                                 const float* __restrict__ wscale,
                                                 uint8_t* __restrict__ Xq,
                                                 float* __restrict__ dq) {
    __shared__ float red[256];
    const int m = blockIdx.x;
    const int t = threadIdx.x;
    const v4f* row = (const v4f*)(X + (size_t)m * K_DIM);
    v4f xv[4];
    float amax = 0.f;
#pragma unroll
    for (int i = 0; i < 4; i++) {
        xv[i] = row[t * 4 + i];
        amax = fmaxf(amax, fmaxf(fmaxf(fabsf(xv[i].x), fabsf(xv[i].y)),
                                 fmaxf(fabsf(xv[i].z), fabsf(xv[i].w))));
    }
    red[t] = amax;
    __syncthreads();
    for (int o = 128; o > 0; o >>= 1) {
        if (t < o) red[t] = fmaxf(red[t], red[t + o]);
        __syncthreads();
    }
    float amaxc = fmaxf(red[0], 1e-5f);
    float scale = 127.0f / amaxc;
    if (t == 0) dq[m] = (*wscale) * amaxc / 127.0f;
    uint32_t pk[4];
#pragma unroll
    for (int i = 0; i < 4; i++) {
        int q0 = (int)fminf(127.f, fmaxf(-128.f, rintf(xv[i].x * scale)));
        int q1 = (int)fminf(127.f, fmaxf(-128.f, rintf(xv[i].y * scale)));
        int q2 = (int)fminf(127.f, fmaxf(-128.f, rintf(xv[i].z * scale)));
        int q3 = (int)fminf(127.f, fmaxf(-128.f, rintf(xv[i].w * scale)));
        pk[i] = (uint32_t)(q0 & 0xFF) | ((uint32_t)(q1 & 0xFF) << 8) |
                ((uint32_t)(q2 & 0xFF) << 16) | ((uint32_t)(q3 & 0xFF) << 24);
    }
    v4i out = {(int)pk[0], (int)pk[1], (int)pk[2], (int)pk[3]};
    ((v4i*)(Xq + (size_t)m * K_DIM))[t] = out;
}

#define LDS_STRIDE 144
#define KSTEP 128
#define TILEB (128 * LDS_STRIDE)

__global__ __launch_bounds__(256) void k_gemm(const uint8_t* __restrict__ Xq,
                                              const uint8_t* __restrict__ Wq,
                                              const float* __restrict__ dq,
                                              const float* __restrict__ bias,
                                              float* __restrict__ Y) {
    __shared__ __align__(16) uint8_t lA[2 * TILEB];
    __shared__ __align__(16) uint8_t lB[2 * TILEB];

    const int t     = threadIdx.x;
    const int lane  = t & 31;
    const int wave  = t >> 5;
    const int waveM = wave & 1;
    const int waveN = wave >> 1;
    const int l16   = lane & 15;
    const int lhalf = lane >> 4;
    const int m0 = blockIdx.y * 128;
    const int n0 = blockIdx.x * 128;

    const uint32_t lAbase = (uint32_t)(uintptr_t)&lA[0];
    const uint32_t lBbase = (uint32_t)(uintptr_t)&lB[0];

    v8i acc[4][2] = {};

    auto issue_stage = [&](int k0, int buf) {
        const uint8_t* gA = Xq + (size_t)m0 * K_DIM + k0;
        const uint8_t* gB = Wq + (size_t)n0 * K_DIM + k0;
        const uint32_t la0 = lAbase + (uint32_t)buf * TILEB;
        const uint32_t lb0 = lBbase + (uint32_t)buf * TILEB;
#pragma unroll
        for (int c = 0; c < 4; c++) {
            int chunk = t + c * 256;
            int row = chunk >> 3;
            int seg = chunk & 7;
            uint64_t ga = (uint64_t)(uintptr_t)(gA + (size_t)row * K_DIM + seg * 16);
            uint32_t la = la0 + (uint32_t)(row * LDS_STRIDE + seg * 16);
            asm volatile("global_load_async_to_lds_b128 %0, %1, off"
                         :: "v"(la), "v"(ga) : "memory");
            uint64_t gb = (uint64_t)(uintptr_t)(gB + (size_t)row * K_DIM + seg * 16);
            uint32_t lb = lb0 + (uint32_t)(row * LDS_STRIDE + seg * 16);
            asm volatile("global_load_async_to_lds_b128 %0, %1, off"
                         :: "v"(lb), "v"(gb) : "memory");
        }
    };

    issue_stage(0, 0);
    asm volatile("s_wait_asynccnt 0x0" ::: "memory");
    __syncthreads();

    int buf = 0;
    for (int k0 = 0; k0 < K_DIM; k0 += KSTEP) {
        if (k0 + KSTEP < K_DIM) issue_stage(k0 + KSTEP, buf ^ 1);

        const uint8_t* bAp = lA + buf * TILEB;
        const uint8_t* bBp = lB + buf * TILEB;
#pragma unroll
        for (int h = 0; h < 2; h++) {
            FragB fb[2];
#pragma unroll
            for (int j = 0; j < 2; j++) {
                const uint8_t* p = bBp + (waveN * 32 + j * 16 + l16) * LDS_STRIDE
                                 + h * 64 + lhalf * 16;
                fb[j].q[0] = *(const v4i*)(p);
                fb[j].q[1] = *(const v4i*)(p + 32);
            }
            FragA fa[4];
#pragma unroll
            for (int i = 0; i < 4; i++) {
                const uint8_t* p = bAp + (waveM * 64 + i * 16 + l16) * LDS_STRIDE
                                 + h * 64 + lhalf * 16;
                fa[i].h[0] = *(const v2i*)(p);
                fa[i].h[1] = *(const v2i*)(p + 8);
                fa[i].h[2] = *(const v2i*)(p + 32);
                fa[i].h[3] = *(const v2i*)(p + 40);
            }
#pragma unroll
            for (int i = 0; i < 4; i++) {
#pragma unroll
                for (int j = 0; j < 2; j++) {
                    acc[i][j] = __builtin_amdgcn_wmma_i32_16x16x64_iu8(
                         true, fa[i].v,  true, fb[j].v,
                        acc[i][j],  false,  false);
                }
            }
        }

        if (k0 + KSTEP < K_DIM) {
            asm volatile("s_wait_asynccnt 0x0" ::: "memory");
            __syncthreads();
            buf ^= 1;
        }
    }

#pragma unroll
    for (int i = 0; i < 4; i++) {
        int mBase = m0 + waveM * 64 + i * 16 + lhalf * 8;
#pragma unroll
        for (int j = 0; j < 2; j++) {
            int n = n0 + waveN * 32 + j * 16 + l16;
            float bn = bias[n];
#pragma unroll
            for (int r = 0; r < 8; r++) {
                int m = mBase + r;
                float val = (float)acc[i][j][r] * dq[m] + bn;
                __builtin_nontemporal_store(val, &Y[(size_t)m * N_DIM + n]);
            }
        }
    }
}

extern "C" void kernel_launch(void* const* d_in, const int* in_sizes, int n_in,
                              void* d_out, int out_size, void* d_ws, size_t ws_size,
                              hipStream_t stream) {
    const float* x    = (const float*)d_in[0];
    const float* w    = (const float*)d_in[1];
    const float* bias = (const float*)d_in[2];
    float* y = (float*)d_out;

    uint8_t* ws      = (uint8_t*)d_ws;
    float*   part    = (float*)(ws + PART_OFF);
    float*   wscale  = (float*)(ws + WSCALE_OFF);
    uint8_t* wq      = ws + WQ_OFF;
    uint8_t* xq      = ws + XQ_OFF;
    float*   dqf     = (float*)(ws + DQ_OFF);

    k_abs_partial<<<1024, 256, 0, stream>>>(w, part);
    k_abs_final<<<1, 256, 0, stream>>>(part, wscale);
    k_quant_w<<<(N_DIM * (size_t)K_DIM / 4) / 256, 256, 0, stream>>>(w, wscale, wq);
    k_quant_x<<<M_DIM, 256, 0, stream>>>(x, wscale, xq, dqf);

    dim3 grid(N_DIM / 128, M_DIM / 128);
    k_gemm<<<grid, 256, 0, stream>>>(xq, wq, dqf, bias, y);
}
